// VQQuantizer_50989851738234
// MI455X (gfx1250) — matrix operand order repaired and checked by reading, not compiled, not run on hardware
//
#include <hip/hip_runtime.h>
#include <hip/hip_bf16.h>
#include <stdint.h>

// ---------------------------------------------------------------------------
// VQ quantize for MI455X (gfx1250, wave32).
//   x:        (32768, 256) f32
//   codebook: (8192, 256)  f32
// Outputs (concat in d_out, all f32):
//   codes (32768) | quantized_st (32768*256) | residuals (32768*256) | loss (1)
//
// GEMM: M=32768 rows, N=8192 codes, K=256 dims, folded argmin epilogue.
// Precision: bf16 hi/lo split, dot = hi*hi + lo*hi + hi*lo via
// v_wmma_f32_16x16x32_bf16; exact f32 ||e||^2 added outside the WMMA.
// Data path: codebook chunks are DMA'd into LDS with CDNA5 async loads
// (global_load_async_to_lds_b128 / ASYNCcnt), de-swizzled into WMMA
// fragment-major layout during the copy, double-buffered, and read back as
// conflict-free ds_load_b128 fragments.
// ---------------------------------------------------------------------------

#define B_ROWS 32768
#define D_DIM  256
#define K_CODES 8192
#define ROW_BYTES 512                 // 256 bf16 per codebook row
#define CHUNK_BYTES 8192              // 16 rows * 512 B (per hi or lo)
#define BUF_BYTES 16384               // hi (8 KB) + lo (8 KB)

#define OFF_CODES 0
#define OFF_QST   ((size_t)B_ROWS)
#define OFF_RES   ((size_t)B_ROWS + (size_t)B_ROWS * D_DIM)
#define OFF_LOSS  ((size_t)B_ROWS + 2ull * (size_t)B_ROWS * D_DIM)

typedef __attribute__((ext_vector_type(16))) __bf16 v16bf;
typedef __attribute__((ext_vector_type(8)))  __bf16 v8bf;
typedef __attribute__((ext_vector_type(8)))  float  v8f;

union Frag16 { v16bf v; __bf16 e[16]; };
union Frag8  { v8bf  v; __bf16 e[8];  };

static __device__ __forceinline__ v8f wmma_bf16(v16bf a, v16bf b, v8f c) {
  // (neg_a, A, neg_b, B, c_mod, C, reuse_a, reuse_b)
  return __builtin_amdgcn_wmma_f32_16x16x32_bf16(false, a, false, b,
                                                 (short)0, c, false, false);
}

// ---------------------------------------------------------------------------
// Kernel 1: codebook f32 -> (hi, lo) bf16 row-major + cnorm[k] = ||e_k||^2
// grid = K_CODES blocks x 32 threads (one wave per code row).
// ---------------------------------------------------------------------------
__global__ void vq_prep_codebook(const float* __restrict__ cb,
                                 __bf16* __restrict__ chi,
                                 __bf16* __restrict__ clo,
                                 float*  __restrict__ cnorm) {
  const int k = blockIdx.x;
  const int lane = threadIdx.x;            // 0..31, 8 elems each
  const float* row = cb + (size_t)k * D_DIM + lane * 8;
  float4 f0 = *(const float4*)(row);
  float4 f1 = *(const float4*)(row + 4);
  float vals[8] = {f0.x, f0.y, f0.z, f0.w, f1.x, f1.y, f1.z, f1.w};

  Frag8 h, l;
  float ss = 0.0f;
#pragma unroll
  for (int j = 0; j < 8; ++j) {
    float v = vals[j];
    __bf16 hb = (__bf16)v;
    float  hv = (float)hb;
    h.e[j] = hb;
    l.e[j] = (__bf16)(v - hv);
    ss += v * v;
  }
  *(v8bf*)(chi + (size_t)k * D_DIM + lane * 8) = h.v;
  *(v8bf*)(clo + (size_t)k * D_DIM + lane * 8) = l.v;

#pragma unroll
  for (int m = 16; m >= 1; m >>= 1) ss += __shfl_xor(ss, m, 32);
  if (lane == 0) cnorm[k] = ss;
}

// ---------------------------------------------------------------------------
// Kernel 2: WMMA GEMM + fused argmin with async-LDS double buffering.
// 128 threads = 4 waves per WG; each wave owns TWO 16-row tiles (A resident
// in VGPRs across the whole N loop); grid = (B_ROWS/16) / (4*2) = 256.
// ---------------------------------------------------------------------------
__global__ void __launch_bounds__(128)
vq_argmin_wmma(const float*  __restrict__ x,
               const __bf16* __restrict__ chi,
               const __bf16* __restrict__ clo,
               const float*  __restrict__ cnorm,
               float*        __restrict__ codes_out) {
  // LDS: 2 buffers, each holds one 16-code chunk in FRAGMENT-MAJOR layout:
  //   fragment f (0..7 = hi kc, 8..15 = lo kc) occupies [f*1024, f*1024+1024)
  //   lane l's 32 bytes (16 bf16 K-values) sit at f*1024 + l*32.
  __shared__ __align__(32) unsigned char ldsb[2][BUF_BYTES];

  const int tid  = threadIdx.x;
  const int lane = tid & 31;
  const int wave = tid >> 5;
  const int gw   = blockIdx.x * 4 + wave;          // global wave index
  const int t0   = gw * 2;                         // first 16-row tile
  const bool lo_half = lane < 16;
  const int abase = lo_half ? 0 : 8;               // A-frag K sub-base per lane

  // --- Precompute this thread's 8 async-copy slots (16 B each, 16 KB/WG) ---
  // dst d = j*2048 + tid*16; decode (fragment, lane, byte) -> static global
  // offset so the LDS image is fragment-major.
  const uint32_t lds_base = (uint32_t)(uintptr_t)(void*)&ldsb[0][0];
  uint32_t dsl[8];
  uint64_t srcb[8];
  {
    const unsigned char* chb = (const unsigned char*)chi;
    const unsigned char* clb = (const unsigned char*)clo;
#pragma unroll
    for (int j = 0; j < 8; ++j) {
      uint32_t d   = (uint32_t)(j * 2048 + tid * 16);
      uint32_t f   = d >> 10;            // fragment 0..15
      uint32_t mat = f >> 3;             // 0 = hi, 1 = lo
      uint32_t kc  = f & 7;
      uint32_t rem = d & 1023;
      uint32_t l   = rem >> 5;           // target lane 0..31
      uint32_t b   = rem & 31;           // 0 or 16
      uint32_t row  = l & 15;            // code within chunk
      uint32_t half = (l >> 4) * 16;     // K-half byte offset
      uint64_t soff = (uint64_t)row * ROW_BYTES + kc * 64 + half + b * 2;
      srcb[j] = (uint64_t)(uintptr_t)(mat ? clb : chb) + soff;
      dsl[j]  = d;
    }
  }

  // Issue async DMA of chunk ci into LDS buffer `buf`.
  auto issue_chunk = [&](int ci, int buf) {
    const uint64_t goff = (uint64_t)ci * CHUNK_BYTES;
    const uint32_t lbase = lds_base + (uint32_t)buf * BUF_BYTES;
#pragma unroll
    for (int j = 0; j < 8; ++j) {
      uint32_t la = lbase + dsl[j];
      uint64_t ga = srcb[j] + goff;
      asm volatile("global_load_async_to_lds_b128 %0, %1, off"
                   :: "v"(la), "v"(ga) : "memory");
    }
  };

  // Kick off chunk 0 DMA, then build A fragments while it flies.
  issue_chunk(0, 0);

  // --- Build resident A fragments (hi/lo split) for both 16-row tiles ---
  Frag16 a_hi[2][8], a_lo[2][8];
#pragma unroll
  for (int m = 0; m < 2; ++m) {
    const int row = (t0 + m) * 16 + (lane & 15);
    const float* xrow = x + (size_t)row * D_DIM;
#pragma unroll
    for (int kc = 0; kc < 8; ++kc) {
      const float* xp = xrow + kc * 32 + abase;
      float4 f0 = *(const float4*)(xp);
      float4 f1 = *(const float4*)(xp + 4);
      float4 f2 = *(const float4*)(xp + 16);
      float4 f3 = *(const float4*)(xp + 20);
      float vals[16] = {f0.x, f0.y, f0.z, f0.w, f1.x, f1.y, f1.z, f1.w,
                        f2.x, f2.y, f2.z, f2.w, f3.x, f3.y, f3.z, f3.w};
#pragma unroll
      for (int j = 0; j < 16; ++j) {
        float v = vals[j];
        __bf16 hb = (__bf16)v;
        a_hi[m][kc].e[j] = hb;
        a_lo[m][kc].e[j] = (__bf16)(v - (float)hb);
      }
    }
  }

  float bestv[2][8];
  int   besti[2][8];
#pragma unroll
  for (int m = 0; m < 2; ++m)
#pragma unroll
    for (int v = 0; v < 8; ++v) { bestv[m][v] = 3.402823466e38f; besti[m][v] = 0; }

  // --- Stream all 8192 codes, 16 per chunk, double-buffered through LDS ---
  for (int ci = 0; ci < K_CODES / 16; ++ci) {
    const int cur = ci & 1;
    // Wait for this wave's outstanding async DMA, then sync the workgroup:
    // after the barrier, buffer `cur` is complete and buffer `cur^1` is free.
    asm volatile("s_wait_asynccnt 0x0" ::: "memory");
    __syncthreads();
    if (ci + 1 < K_CODES / 16) issue_chunk(ci + 1, cur ^ 1);

    const unsigned char* lb = &ldsb[cur][0];
    v8f acc0 = {};
    v8f acc1 = {};
#pragma unroll
    for (int kc = 0; kc < 8; ++kc) {
      v16bf bh = *(const v16bf*)(lb + (kc    ) * 1024 + lane * 32);
      v16bf bl = *(const v16bf*)(lb + (kc + 8) * 1024 + lane * 32);
      acc0 = wmma_bf16(a_hi[0][kc].v, bh, acc0);   // hi*hi  (tile 0)
      acc1 = wmma_bf16(a_hi[1][kc].v, bh, acc1);   // hi*hi  (tile 1)
      acc0 = wmma_bf16(a_lo[0][kc].v, bh, acc0);   // lo*hi
      acc1 = wmma_bf16(a_lo[1][kc].v, bh, acc1);
      acc0 = wmma_bf16(a_hi[0][kc].v, bl, acc0);   // hi*lo
      acc1 = wmma_bf16(a_hi[1][kc].v, bl, acc1);
    }

    const int c = ci * 16 + (lane & 15);           // this lane's column/code
    const float cn = cnorm[c];
#pragma unroll
    for (int v = 0; v < 8; ++v) {
      float s0 = fmaf(-2.0f, acc0[v], cn);         // -2 x.e + ||e||^2
      float s1 = fmaf(-2.0f, acc1[v], cn);
      if (s0 < bestv[0][v]) { bestv[0][v] = s0; besti[0][v] = c; }
      if (s1 < bestv[1][v]) { bestv[1][v] = s1; besti[1][v] = c; }
    }
  }

  // --- Reduce across the 16 columns (lanes within each 16-lane half). ---
#pragma unroll
  for (int m = 0; m < 2; ++m) {
#pragma unroll
    for (int v = 0; v < 8; ++v) {
      float bv = bestv[m][v];
      int   bi = besti[m][v];
#pragma unroll
      for (int s = 8; s >= 1; s >>= 1) {
        float ov = __shfl_xor(bv, s, 32);
        int   oi = __shfl_xor(bi, s, 32);
        if (ov < bv || (ov == bv && oi < bi)) { bv = ov; bi = oi; }
      }
      bestv[m][v] = bv; besti[m][v] = bi;
    }
    if ((lane & 15) == 0) {
      const int rb = (t0 + m) * 16 + (lo_half ? 0 : 8);
#pragma unroll
      for (int v = 0; v < 8; ++v) codes_out[rb + v] = (float)besti[m][v];
    }
  }
}

// ---------------------------------------------------------------------------
// Kernel 3: gather e_k, write quantized_st / residuals, per-block loss partial.
// 256 threads x 8 elems = 2048 elems (8 rows) per block; grid = 4096.
// ---------------------------------------------------------------------------
__global__ void __launch_bounds__(256)
vq_gather_residual(const float* __restrict__ x,
                   const float* __restrict__ cb,
                   float*       __restrict__ out,
                   float*       __restrict__ partials) {
  __shared__ float red[256];
  const int tid = threadIdx.x;
  const size_t e0 = (size_t)blockIdx.x * 2048 + (size_t)tid * 8;
  const int row = (int)(e0 >> 8);
  const int col = (int)(e0 & 255);

  const int code = (int)out[OFF_CODES + row];
  const float* xp = x + e0;
  const float* ep = cb + (size_t)code * D_DIM + col;

  float4 xa = *(const float4*)(xp);
  float4 xb = *(const float4*)(xp + 4);
  float4 ea = *(const float4*)(ep);
  float4 eb = *(const float4*)(ep + 4);

  // quantized_st value == e_k (straight-through only affects gradients)
  *(float4*)(out + OFF_QST + e0)     = ea;
  *(float4*)(out + OFF_QST + e0 + 4) = eb;

  float4 ra = {xa.x - ea.x, xa.y - ea.y, xa.z - ea.z, xa.w - ea.w};
  float4 rb = {xb.x - eb.x, xb.y - eb.y, xb.z - eb.z, xb.w - eb.w};
  *(float4*)(out + OFF_RES + e0)     = ra;
  *(float4*)(out + OFF_RES + e0 + 4) = rb;

  float s = ra.x * ra.x + ra.y * ra.y + ra.z * ra.z + ra.w * ra.w +
            rb.x * rb.x + rb.y * rb.y + rb.z * rb.z + rb.w * rb.w;

  red[tid] = s;
  __syncthreads();
  for (int m = 128; m >= 1; m >>= 1) {
    if (tid < m) red[tid] += red[tid + m];
    __syncthreads();
  }
  if (tid == 0) partials[blockIdx.x] = red[0];
}

// ---------------------------------------------------------------------------
// Kernel 4: deterministic final reduction -> commitment loss.
// ---------------------------------------------------------------------------
__global__ void __launch_bounds__(256)
vq_loss_reduce(const float* __restrict__ partials, float* __restrict__ out) {
  __shared__ float red[256];
  const int tid = threadIdx.x;
  float s = 0.0f;
  for (int j = 0; j < 16; ++j) s += partials[tid + 256 * j];  // fixed order
  red[tid] = s;
  __syncthreads();
  for (int m = 128; m >= 1; m >>= 1) {
    if (tid < m) red[tid] += red[tid + m];
    __syncthreads();
  }
  if (tid == 0)
    out[OFF_LOSS] = 0.25f * red[0] / (float)((size_t)B_ROWS * D_DIM);
}

// ---------------------------------------------------------------------------
extern "C" void kernel_launch(void* const* d_in, const int* in_sizes, int n_in,
                              void* d_out, int out_size, void* d_ws, size_t ws_size,
                              hipStream_t stream) {
  const float* x  = (const float*)d_in[0];  // (32768, 256)
  const float* cb = (const float*)d_in[1];  // (8192, 256)
  float* out = (float*)d_out;

  // Workspace carve-up (~8.4 MB total)
  __bf16* chi = (__bf16*)d_ws;                         // 8192*256 bf16
  __bf16* clo = chi + (size_t)K_CODES * D_DIM;         // 8192*256 bf16
  float* cnorm    = (float*)(clo + (size_t)K_CODES * D_DIM); // 8192 f32
  float* partials = cnorm + K_CODES;                   // 4096 f32

  vq_prep_codebook<<<K_CODES, 32, 0, stream>>>(cb, chi, clo, cnorm);

  // (B_ROWS/16) tiles / (4 waves * 2 tiles) per block = 256 blocks
  vq_argmin_wmma<<<(B_ROWS / 16) / 8, 128, 0, stream>>>(x, chi, clo, cnorm,
                                                        out + OFF_CODES);

  vq_gather_residual<<<(B_ROWS * D_DIM) / 2048, 256, 0, stream>>>(x, cb, out,
                                                                  partials);

  vq_loss_reduce<<<1, 256, 0, stream>>>(partials, out);
}
